// AttentionRoutingModel_89343909692186
// MI455X (gfx1250) — matrix operand order repaired and checked by reading, not compiled, not run on hardware
//
#include <hip/hip_runtime.h>

// ---------------------------------------------------------------------------
// CDNA5 (gfx1250) implementation of the attention-routing model forward pass.
// All large matmuls (conv2 implicit GEMM, big/small MLPs, aggregator, head L1)
// run on v_wmma_f32_16x16x32_f16 (wave32). The GEMM stages its shared B tile
// in LDS (double-buffered), preferring GLOBAL_LOAD_ASYNC_TO_LDS_B128 +
// s_wait_asynccnt when the toolchain exposes the builtins.
// ---------------------------------------------------------------------------

typedef __attribute__((ext_vector_type(16))) _Float16 v16h;
typedef __attribute__((ext_vector_type(8)))  _Float16 v8h;
typedef __attribute__((ext_vector_type(8)))  float    v8f;
typedef __attribute__((ext_vector_type(4)))  int      v4i;

union F8 { v8f v; float f[8]; };

static __device__ __forceinline__ v16h pack16(v8h a, v8h b) {
  return __builtin_shufflevector(a, b, 0,1,2,3,4,5,6,7,8,9,10,11,12,13,14,15);
}

static __device__ __forceinline__ v8f wmma_f16(v16h a, v16h b, v8f c) {
  // (neg_a, A, neg_b, B, c_mod, C, reuse_a, reuse_b)
  return __builtin_amdgcn_wmma_f32_16x16x32_f16(false, a, false, b, (short)0, c,
                                                false, false);
}

#if defined(__HIP_DEVICE_COMPILE__) && \
    __has_builtin(__builtin_amdgcn_global_load_async_to_lds_b128) && \
    __has_builtin(__builtin_amdgcn_s_wait_asynccnt)
#define GEMM_ASYNC 1
#else
#define GEMM_ASYNC 0
#endif

#if GEMM_ASYNC
// builtin signature (from clang diagnostic): (v4i AS1*, v4i AS3*, imm offset, imm cpol)
#define ASYNC_CP_B128(gptr, lptr)                                      \
  __builtin_amdgcn_global_load_async_to_lds_b128(                      \
      (__attribute__((address_space(1))) v4i*)(gptr),                  \
      (__attribute__((address_space(3))) v4i*)(lptr), 0, 0)
#endif

// ---------------------------------------------------------------------------
// prep kernels
// ---------------------------------------------------------------------------

__global__ void cvt_f32_f16(const float* __restrict__ in, _Float16* __restrict__ out, int n) {
  int i = blockIdx.x * 256 + threadIdx.x;
  if (i < n) out[i] = (_Float16)in[i];
}

// W (O x K) f32 -> WB (K x O) f16, WB[k*O+o] = W[o*K+k]
__global__ void xpose_f32_f16(const float* __restrict__ in, _Float16* __restrict__ out,
                              int O, int K) {
  int idx = blockIdx.x * 256 + threadIdx.x;
  if (idx >= O * K) return;
  int k = idx / O, o = idx - k * O;
  out[idx] = (_Float16)in[(size_t)o * K + k];
}

// cw2 (co,ci,ky,kx)=(64,64,3,3) f32 -> WC2[(tap*64+ci)*64+co] f16, tap=ky*3+kx
__global__ void reorg_cw2(const float* __restrict__ w, _Float16* __restrict__ out) {
  int idx = blockIdx.x * 256 + threadIdx.x;   // 576*64
  if (idx >= 576 * 64) return;
  int k = idx >> 6, co = idx & 63;
  int tap = k >> 6, ci = k & 63;
  out[idx] = (_Float16)w[co * 576 + ci * 9 + tap];
}

// ---------------------------------------------------------------------------
// conv1: (512,3,128,128) f32 NCHW -> relu -> f1 (512,64,64,64) NHWC f16
// stride 2, 3x3, SAME (pad_before=0, pad_after=1). One thread = one pixel,
// all 64 output channels; weights staged in LDS.
// ---------------------------------------------------------------------------
__global__ void conv1_kernel(const float* __restrict__ img,
                             const float* __restrict__ w,     // (64,3,3,3)
                             const float* __restrict__ bias,  // (64)
                             _Float16* __restrict__ f1) {
  __shared__ float sw[64 * 27];
  __shared__ float sb[64];
  for (int i = threadIdx.x; i < 64 * 27; i += 256) sw[i] = w[i];
  if (threadIdx.x < 64) sb[threadIdx.x] = bias[threadIdx.x];
  __syncthreads();

  int b   = blockIdx.x >> 4;                        // 16 blocks / image
  int pix = ((blockIdx.x & 15) << 8) + threadIdx.x; // 0..4095
  int y = pix >> 6, x = pix & 63;
  const float* ib = img + (size_t)b * 3 * 128 * 128;

  float v[27];
#pragma unroll
  for (int c = 0; c < 3; ++c)
#pragma unroll
    for (int ky = 0; ky < 3; ++ky)
#pragma unroll
      for (int kx = 0; kx < 3; ++kx) {
        int iy = 2 * y + ky, ix = 2 * x + kx;
        v[c * 9 + ky * 3 + kx] =
            (iy < 128 && ix < 128) ? ib[c * 16384 + iy * 128 + ix] : 0.f;
      }

  _Float16* op = f1 + ((size_t)b * 4096 + pix) * 64;
  for (int co = 0; co < 64; ++co) {
    float s = sb[co];
    const float* wr = &sw[co * 27];
#pragma unroll
    for (int i = 0; i < 27; ++i) s = fmaf(v[i], wr[i], s);
    op[co] = (_Float16)fmaxf(s, 0.f);
  }
}

// ---------------------------------------------------------------------------
// conv2 implicit GEMM + relu + 8x8 mean pool, fused. Never materializes f2.
// Per block: one image slice of 128 pixels (4 rows of 32) x all 64 channels.
// M=1024 pixels/img, N=64, K=576=(9 taps x 64 ci). WMMA 16x16x32 f16.
// Pool partials reduced in LDS, then atomically added into tokens (B,16,64).
// ---------------------------------------------------------------------------
__global__ void conv2_wmma_kernel(const _Float16* __restrict__ f1,   // (B,64,64,64) NHWC
                                  const _Float16* __restrict__ wb,   // (576,64) K-major
                                  const float* __restrict__ bias,    // (64)
                                  float* __restrict__ tokens) {      // (B,16,64)
  __shared__ float lacc[4 * 64];   // 4 px-patches x 64 channels
  lacc[threadIdx.x] = 0.f;
  __syncthreads();

  int b  = blockIdx.x >> 3;   // 8 blocks / image
  int pb = blockIdx.x & 7;    // pixel-block: rows y0=pb*4 .. +3 (never crosses py)
  int wave = threadIdx.x >> 5;
  int lane = threadIdx.x & 31;
  int row = lane & 15, hi = lane >> 4, kb = hi * 8;

  int m = pb * 128 + wave * 16 + row;   // pixel for this A-fragment row
  int y = m >> 5, x = m & 31;
  const _Float16* fb = f1 + (size_t)b * 64 * 64 * 64;

  v8f acc[4] = {};
  for (int ks = 0; ks < 576; ks += 32) {
    v8h lo = {}, hi8 = {};
    {
      int k0 = ks + kb;                       // run of 8: single tap
      int tap = k0 >> 6, ci = k0 & 63;
      int ky = tap / 3, kx = tap % 3;
      int iy = 2 * y + ky, ix = 2 * x + kx;   // pad_before = 0
      if (iy < 64 && ix < 64)
        lo = *(const v8h*)(fb + ((size_t)((iy << 6) + ix) << 6) + ci);
    }
    {
      int k1 = ks + kb + 16;
      int tap = k1 >> 6, ci = k1 & 63;
      int ky = tap / 3, kx = tap % 3;
      int iy = 2 * y + ky, ix = 2 * x + kx;
      if (iy < 64 && ix < 64)
        hi8 = *(const v8h*)(fb + ((size_t)((iy << 6) + ix) << 6) + ci);
    }
    v16h af = pack16(lo, hi8);

    const _Float16* bcol = wb + (size_t)(ks + kb) * 64 + row;   // N col = lane%16, K rows kb.., kb+16..
#pragma unroll
    for (int nt = 0; nt < 4; ++nt) {
      v16h bf;
#pragma unroll
      for (int s = 0; s < 8; ++s) {
        bf[s] = bcol[s * 64 + nt * 16];
        bf[8 + s] = bcol[(16 + s) * 64 + nt * 16];
      }
      acc[nt] = wmma_f16(af, bf, acc[nt]);
    }
  }

  // epilogue: bias + relu + pooled accumulation into LDS
  int mbase = pb * 128 + wave * 16;
#pragma unroll
  for (int nt = 0; nt < 4; ++nt) {
    int c = nt * 16 + (lane & 15);
    float bv = bias[c];
    F8 u; u.v = acc[nt];
#pragma unroll
    for (int r = 0; r < 8; ++r) {
      int pm = mbase + r + 8 * hi;          // pixel index (C-layout M)
      int px = (pm & 31) >> 3;              // x/8
      float val = fmaxf(u.f[r] + bv, 0.f);
      atomicAdd(&lacc[px * 64 + c], val);
    }
  }
  __syncthreads();

  // 256 token partials (4 px-patches x 64 ch) -> global atomics, /64 for mean
  {
    int px = threadIdx.x >> 6;
    int c  = threadIdx.x & 63;
    int py = pb >> 1;                       // y0=pb*4 -> py = pb/2
    int p  = py * 4 + px;
    atomicAdd(&tokens[((size_t)b * 16 + p) * 64 + c],
              lacc[px * 64 + c] * (1.0f / 64.0f));
  }
}

// ---------------------------------------------------------------------------
// attention MLP (64->32->1) + sigmoid + hard threshold -> mask (B*16) f32.
// Forward value of the straight-through router is the hard mask.
// ---------------------------------------------------------------------------
__global__ void attn_mask_kernel(const float* __restrict__ tokens,
                                 const float* __restrict__ aw1, const float* __restrict__ ab1,
                                 const float* __restrict__ aw2, const float* __restrict__ ab2,
                                 const float* __restrict__ thr,
                                 float* __restrict__ mask) {
  int t = blockIdx.x * 256 + threadIdx.x;
  if (t >= 512 * 16) return;
  const float* tok = tokens + (size_t)t * 64;
  float s = ab2[0];
  for (int j = 0; j < 32; ++j) {
    float a = ab1[j];
    const float* w = aw1 + j * 64;
#pragma unroll 8
    for (int i = 0; i < 64; ++i) a = fmaf(w[i], tok[i], a);
    s = fmaf(fmaxf(a, 0.f), aw2[j], s);
  }
  float score = 1.f / (1.f + __expf(-s));
  mask[t] = (score > thr[0]) ? 1.f : 0.f;
}

// ---------------------------------------------------------------------------
// Generic WMMA GEMM: out = [relu](A(MxK,f16) @ Bw(KxN,f16 K-major) + bias).
// Block = 8 waves; wave -> 2 M-tiles (32 rows) x 64 cols (8 accumulators).
// Block tile = 256 rows x 64 cols; grid = (M/256, N/64).
// Shared 32x64 B tile double-buffered in LDS (rows padded to 72 halves to
// spread ds_load_b128 banks); staged with async-to-LDS when available.
// ---------------------------------------------------------------------------
__global__ __launch_bounds__(256)
void gemm_wmma(const _Float16* __restrict__ A,
               const _Float16* __restrict__ Bw,
               const float* __restrict__ bias,
               float* __restrict__ outF,
               _Float16* __restrict__ outH,
               int M, int N, int K, int relu) {
  __shared__ _Float16 sB[2][32 * 72];    // 2 x 4.5KB

  int tid = threadIdx.x;
  int wave = tid >> 5, lane = tid & 31;
  int m0 = (blockIdx.x * 8 + wave) * 32;     // 2 M-tiles per wave
  int n0 = blockIdx.y * 64;
  int row = lane & 15, hi = lane >> 4, kb = hi * 8;

  // B staging: thread copies one 16B chunk: row kr (0..31), col chunk cc
  int kr = tid >> 3, cc = (tid & 7) * 8;
  const _Float16* gB = Bw + (size_t)kr * N + n0 + cc;
  _Float16* lB[2] = { &sB[0][kr * 72 + cc], &sB[1][kr * 72 + cc] };

  v8f acc[8] = {};
  const _Float16* arow0 = A + (size_t)(m0 + row) * K + kb;
  const _Float16* arow1 = arow0 + (size_t)16 * K;

  const int nk = K >> 5;
  // prologue: stage K-step 0 into buffer 0
#if GEMM_ASYNC
  ASYNC_CP_B128(gB, lB[0]);
#else
  *(v8h*)lB[0] = *(const v8h*)gB;
#endif

  for (int i = 0; i < nk; ++i) {
    const int ks = i << 5;
#if GEMM_ASYNC
    __builtin_amdgcn_s_wait_asynccnt(0);   // my staged chunk has landed
#endif
    __syncthreads();                       // everyone's chunk landed; prior
                                           // buffer reads fully drained
    if (i + 1 < nk) {
      const _Float16* gnext = gB + ((size_t)(i + 1) << 5) * N;
      _Float16* lnext = lB[(i + 1) & 1];
#if GEMM_ASYNC
      ASYNC_CP_B128(gnext, lnext);
#else
      *(v8h*)lnext = *(const v8h*)gnext;
#endif
    }

    // A fragments for the two M tiles (two 16B runs each)
    v16h a0 = pack16(*(const v8h*)(arow0 + ks), *(const v8h*)(arow0 + ks + 16));
    v16h a1 = pack16(*(const v8h*)(arow1 + ks), *(const v8h*)(arow1 + ks + 16));

    const _Float16* bcol = &sB[i & 1][kb * 72 + row];   // B col N = lane%16, K rows kb.., kb+16..
#pragma unroll
    for (int nt = 0; nt < 4; ++nt) {
      v16h bf;
#pragma unroll
      for (int s = 0; s < 8; ++s) {
        bf[s] = bcol[s * 72 + nt * 16];
        bf[8 + s] = bcol[(16 + s) * 72 + nt * 16];
      }
      acc[nt]     = wmma_f16(a0, bf, acc[nt]);
      acc[4 + nt] = wmma_f16(a1, bf, acc[4 + nt]);
    }
  }

  if (m0 >= M) return;   // wave-uniform (all launches have M % 256 == 0)

#pragma unroll
  for (int mt = 0; mt < 2; ++mt) {
#pragma unroll
    for (int nt = 0; nt < 4; ++nt) {
      int c = n0 + nt * 16 + (lane & 15);
      float bv = bias ? bias[c] : 0.f;
      F8 u; u.v = acc[mt * 4 + nt];
#pragma unroll
      for (int r = 0; r < 8; ++r) {
        float v = u.f[r] + bv;
        if (relu) v = fmaxf(v, 0.f);
        size_t o = (size_t)(m0 + mt * 16 + r + 8 * hi) * N + c;
        if (outF) outF[o] = v;
        if (outH) outH[o] = (_Float16)v;
      }
    }
  }
}

// combined = big*mask + small*(1-mask) -> f16 (8192 x 128)
__global__ void combine_kernel(const float* __restrict__ big,
                               const float* __restrict__ small,
                               const float* __restrict__ mask,
                               _Float16* __restrict__ comb) {
  int idx = blockIdx.x * 256 + threadIdx.x;
  if (idx >= 8192 * 128) return;
  float m = mask[idx >> 7];
  comb[idx] = (_Float16)(big[idx] * m + small[idx] * (1.f - m));
}

// final head: logits(512,10) = t1(512,128 f16) @ hw2(10,128).T + hb2  (f32)
__global__ void head_kernel(const _Float16* __restrict__ t1,
                            const float* __restrict__ hw2,
                            const float* __restrict__ hb2,
                            float* __restrict__ out) {
  int idx = blockIdx.x * 256 + threadIdx.x;
  if (idx >= 512 * 10) return;
  int b = idx / 10, o = idx - b * 10;
  const _Float16* r = t1 + (size_t)b * 128;
  const float* w = hw2 + o * 128;
  float s = hb2[o];
#pragma unroll 8
  for (int i = 0; i < 128; ++i) s = fmaf((float)r[i], w[i], s);
  out[idx] = s;
}

// ---------------------------------------------------------------------------
extern "C" void kernel_launch(void* const* d_in, const int* in_sizes, int n_in,
                              void* d_out, int out_size, void* d_ws, size_t ws_size,
                              hipStream_t stream) {
  (void)in_sizes; (void)n_in; (void)out_size; (void)ws_size;
  const float* images = (const float*)d_in[0];
  const float* patches = (const float*)d_in[1];
  const float* cw1 = (const float*)d_in[2];
  const float* cb1 = (const float*)d_in[3];
  const float* cw2 = (const float*)d_in[4];
  const float* cb2 = (const float*)d_in[5];
  const float* aw1 = (const float*)d_in[6];
  const float* ab1 = (const float*)d_in[7];
  const float* aw2 = (const float*)d_in[8];
  const float* ab2 = (const float*)d_in[9];
  const float* thr = (const float*)d_in[10];
  const float* bw1 = (const float*)d_in[11]; const float* bb1 = (const float*)d_in[12];
  const float* bw2 = (const float*)d_in[13]; const float* bb2 = (const float*)d_in[14];
  const float* bw3 = (const float*)d_in[15]; const float* bb3 = (const float*)d_in[16];
  const float* bwo = (const float*)d_in[17]; const float* bbo = (const float*)d_in[18];
  const float* sw1 = (const float*)d_in[19]; const float* sb1 = (const float*)d_in[20];
  const float* swo = (const float*)d_in[21]; const float* sbo = (const float*)d_in[22];
  const float* gw  = (const float*)d_in[23]; const float* gb  = (const float*)d_in[24];
  const float* hw1 = (const float*)d_in[25]; const float* hb1 = (const float*)d_in[26];
  const float* hw2 = (const float*)d_in[27]; const float* hb2 = (const float*)d_in[28];

  size_t off = 0;
  auto alloc = [&](size_t bytes) -> void* {
    off = (off + 255) & ~(size_t)255;
    void* p = (char*)d_ws + off;
    off += bytes;
    return p;
  };

  _Float16* f1    = (_Float16*)alloc((size_t)512 * 64 * 64 * 64 * 2); // conv1 out NHWC
  _Float16* hA    = (_Float16*)alloc((size_t)8192 * 3072 * 2);        // patches f16
  _Float16* h1    = (_Float16*)alloc((size_t)8192 * 512 * 2);
  _Float16* h2    = (_Float16*)alloc((size_t)8192 * 512 * 2);
  _Float16* h3    = (_Float16*)alloc((size_t)8192 * 512 * 2);
  _Float16* s1    = (_Float16*)alloc((size_t)8192 * 64 * 2);
  float*    bigo  = (float*)   alloc((size_t)8192 * 128 * 4);
  float*    smlo  = (float*)   alloc((size_t)8192 * 128 * 4);
  _Float16* comb  = (_Float16*)alloc((size_t)8192 * 128 * 2);
  float*    toks  = (float*)   alloc((size_t)512 * 16 * 64 * 4);
  float*    mask  = (float*)   alloc((size_t)512 * 16 * 4);
  _Float16* g16   = (_Float16*)alloc((size_t)512 * 256 * 2);
  _Float16* t1h   = (_Float16*)alloc((size_t)512 * 128 * 2);
  _Float16* WB1   = (_Float16*)alloc((size_t)3072 * 512 * 2);
  _Float16* WB2   = (_Float16*)alloc((size_t)512 * 512 * 2);
  _Float16* WB3   = (_Float16*)alloc((size_t)512 * 512 * 2);
  _Float16* WBo   = (_Float16*)alloc((size_t)512 * 128 * 2);
  _Float16* WS1   = (_Float16*)alloc((size_t)3072 * 64 * 2);
  _Float16* WSo   = (_Float16*)alloc((size_t)64 * 128 * 2);
  _Float16* WG    = (_Float16*)alloc((size_t)2048 * 256 * 2);
  _Float16* WH1   = (_Float16*)alloc((size_t)256 * 128 * 2);
  _Float16* WC2   = (_Float16*)alloc((size_t)576 * 64 * 2);

  // token accumulator must be zeroed every call (atomically accumulated)
  (void)hipMemsetAsync(toks, 0, (size_t)512 * 16 * 64 * 4, stream);

  // weight prep (K-major f16)
  xpose_f32_f16<<<(512 * 3072 + 255) / 256, 256, 0, stream>>>(bw1, WB1, 512, 3072);
  xpose_f32_f16<<<(512 * 512 + 255) / 256, 256, 0, stream>>>(bw2, WB2, 512, 512);
  xpose_f32_f16<<<(512 * 512 + 255) / 256, 256, 0, stream>>>(bw3, WB3, 512, 512);
  xpose_f32_f16<<<(128 * 512 + 255) / 256, 256, 0, stream>>>(bwo, WBo, 128, 512);
  xpose_f32_f16<<<(64 * 3072 + 255) / 256, 256, 0, stream>>>(sw1, WS1, 64, 3072);
  xpose_f32_f16<<<(128 * 64 + 255) / 256, 256, 0, stream>>>(swo, WSo, 128, 64);
  xpose_f32_f16<<<(256 * 2048 + 255) / 256, 256, 0, stream>>>(gw, WG, 256, 2048);
  xpose_f32_f16<<<(128 * 256 + 255) / 256, 256, 0, stream>>>(hw1, WH1, 128, 256);
  reorg_cw2<<<(576 * 64 + 255) / 256, 256, 0, stream>>>(cw2, WC2);

  // activations
  cvt_f32_f16<<<(8192 * 3072 + 255) / 256, 256, 0, stream>>>(patches, hA, 8192 * 3072);

  // conv branch -> tokens -> mask
  conv1_kernel<<<512 * 16, 256, 0, stream>>>(images, cw1, cb1, f1);
  conv2_wmma_kernel<<<512 * 8, 256, 0, stream>>>(f1, WC2, cb2, toks);
  attn_mask_kernel<<<(8192 + 255) / 256, 256, 0, stream>>>(toks, aw1, ab1, aw2, ab2, thr, mask);

  // big MLP chain (WMMA)  grid = (M/256, N/64)
  gemm_wmma<<<dim3(32, 8), 256, 0, stream>>>(hA, WB1, bb1, nullptr, h1, 8192, 512, 3072, 1);
  gemm_wmma<<<dim3(32, 8), 256, 0, stream>>>(h1, WB2, bb2, nullptr, h2, 8192, 512, 512, 1);
  gemm_wmma<<<dim3(32, 8), 256, 0, stream>>>(h2, WB3, bb3, nullptr, h3, 8192, 512, 512, 1);
  gemm_wmma<<<dim3(32, 2), 256, 0, stream>>>(h3, WBo, bbo, bigo, nullptr, 8192, 128, 512, 0);

  // small MLP chain (WMMA)
  gemm_wmma<<<dim3(32, 1), 256, 0, stream>>>(hA, WS1, sb1, nullptr, s1, 8192, 64, 3072, 1);
  gemm_wmma<<<dim3(32, 2), 256, 0, stream>>>(s1, WSo, sbo, smlo, nullptr, 8192, 128, 64, 0);

  // route + aggregate + head
  combine_kernel<<<(8192 * 128 + 255) / 256, 256, 0, stream>>>(bigo, smlo, mask, comb);
  gemm_wmma<<<dim3(2, 4), 256, 0, stream>>>(comb, WG, gb, nullptr, g16, 512, 256, 2048, 0);
  gemm_wmma<<<dim3(2, 2), 256, 0, stream>>>(g16, WH1, hb1, nullptr, t1h, 512, 128, 256, 1);
  head_kernel<<<(5120 + 255) / 256, 256, 0, stream>>>(t1h, hw2, hb2, (float*)d_out);
}
